// MultiHeadAttention_14104672600194
// MI455X (gfx1250) — matrix operand order repaired and checked by reading, not compiled, not run on hardware
//
#include <hip/hip_runtime.h>
#include <hip/hip_bf16.h>
#include <math.h>

// ---------------------------------------------------------------------------
// Multi-head attention forward, bf16 WMMA pipeline for gfx1250 (MI455X).
//   B=8, S=1024, E=512, H=8, dh=64, KEY_DIM=VALUE_DIM=512.
// Pipeline:
//   1) wconv:   Wq/Wk/Wv/Wo  f32[E,512] -> bf16 transposed Wt[n][k]
//   2) proj:    Q = X@Wq+b (bf16 row-major), K likewise, V stored transposed
//               per head as Vt[B,H,dh,S] (bf16).  16x64 output per wave
//               (A-fragment reused across 4 WMMAs per k-step).
//   3) attn:    per (b,h,16-row q block): scores strip in LDS (f32),
//               softmax (scale=1/sqrt(512)), P@V via WMMA, write merged-head
//               O bf16 [B,S,512]
//   4) outproj: out = O@Wo + bo  (f32, written to d_out), 16x64 per wave
// All matmuls use v_wmma_f32_16x16x32_bf16 (f32 accumulation).
// ---------------------------------------------------------------------------

typedef __attribute__((ext_vector_type(16))) __bf16 v16bf;
typedef __attribute__((ext_vector_type(8)))  float  v8f;

#define SROW 1028   // LDS score row stride in floats (1024 + 4 pad -> conflict-free)

__device__ inline v8f wmma_bf16(v16bf a, v16bf b, v8f c) {
  return __builtin_amdgcn_wmma_f32_16x16x32_bf16(
      /*neg_a=*/false, a, /*neg_b=*/false, b,
      /*c_mod=*/(short)0, c, /*reuse_a=*/false, /*reuse_b=*/false);
}

// A-fragment (16x32, bf16): lane holds row (lane&15); K = half*8 + {0..7,16..23}
__device__ inline v16bf load_a_bf16(const __bf16* __restrict__ p, int half) {
  const __bf16* q = p + half * 8;
  v16bf v;
#pragma unroll
  for (int i = 0; i < 8; ++i) v[i] = q[i];
#pragma unroll
  for (int i = 0; i < 8; ++i) v[8 + i] = q[16 + i];
  return v;
}

// A-fragment from f32 source (convert to bf16 on the fly)
__device__ inline v16bf load_a_f32(const float* __restrict__ p, int half) {
  const float* q = p + half * 8;
  v16bf v;
#pragma unroll
  for (int i = 0; i < 8; ++i) v[i] = (__bf16)q[i];
#pragma unroll
  for (int i = 0; i < 8; ++i) v[8 + i] = (__bf16)q[16 + i];
  return v;
}

// B-fragment (32x16, bf16) from B^T-row-major storage:
// lane holds column (lane&15); K = half*8 + {0..7,16..23}
__device__ inline v16bf load_b_bf16(const __bf16* __restrict__ p, int half) {
  const __bf16* q = p + half * 8;
  v16bf v;
#pragma unroll
  for (int i = 0; i < 8; ++i) v[i] = q[i];
#pragma unroll
  for (int i = 0; i < 8; ++i) v[8 + i] = q[16 + i];
  return v;
}

// ---------------------------------------------------------------------------
// 1) Weight convert + transpose: Wt[n*512 + k] = (bf16) W[k*512 + n]
// ---------------------------------------------------------------------------
__global__ void wconv_kernel(const float* __restrict__ W, __bf16* __restrict__ Wt) {
  int idx = blockIdx.x * 256 + threadIdx.x;   // 0 .. 512*512-1
  int n = idx >> 9, k = idx & 511;
  Wt[(size_t)n * 512 + k] = (__bf16)W[(size_t)k * 512 + n];
}

// ---------------------------------------------------------------------------
// 2) Projection GEMM: C[8192x512] = A(f32) @ W + bias, A row-major.
//    Each wave computes a 16x64 slab (4 n-tiles); A fragment reused 4x.
//    mode 0: store bf16 row-major.  mode 1: store V transposed per head.
// ---------------------------------------------------------------------------
__global__ __launch_bounds__(128) void proj_kernel(
    const float* __restrict__ A, const __bf16* __restrict__ Wt,
    const float* __restrict__ bias, __bf16* __restrict__ C, int mode) {
  int wv = threadIdx.x >> 5, ln = threadIdx.x & 31;
  int l15 = ln & 15, half = ln >> 4;
  int slab = blockIdx.x * 4 + wv;       // 512 row-tiles x 8 col-groups = 4096
  int mt = slab >> 3, ng = slab & 7;
  int m0 = mt * 16, n0 = ng * 64;

  const float* Arow = A + (size_t)(m0 + l15) * 512;
  const __bf16* Brow[4];
  v8f acc[4];
#pragma unroll
  for (int j = 0; j < 4; ++j) {
    Brow[j] = Wt + (size_t)(n0 + j * 16 + l15) * 512;
    float bval = bias[n0 + j * 16 + l15];
#pragma unroll
    for (int r = 0; r < 8; ++r) acc[j][r] = bval;
  }

  for (int k0 = 0; k0 < 512; k0 += 32) {
    if (k0 + 32 < 512) {
      __builtin_prefetch(Arow + k0 + 32, 0, 0);
#pragma unroll
      for (int j = 0; j < 4; ++j) __builtin_prefetch(Brow[j] + k0 + 32, 0, 0);
    }
    v16bf a = load_a_f32(Arow + k0, half);
#pragma unroll
    for (int j = 0; j < 4; ++j) {
      v16bf b = load_b_bf16(Brow[j] + k0, half);
      acc[j] = wmma_bf16(a, b, acc[j]);
    }
  }

  if (mode == 0) {
#pragma unroll
    for (int j = 0; j < 4; ++j)
#pragma unroll
      for (int r = 0; r < 8; ++r) {
        int m = m0 + r + half * 8;
        C[(size_t)m * 512 + n0 + j * 16 + l15] = (__bf16)acc[j][r];
      }
  } else {
    // V transposed per head: Vt[((b*8+h)*64+d)*1024 + s]
#pragma unroll
    for (int j = 0; j < 4; ++j)
#pragma unroll
      for (int r = 0; r < 8; ++r) {
        int m = m0 + r + half * 8;
        int n = n0 + j * 16 + l15;
        int b_ = m >> 10, s_ = m & 1023;
        int h_ = n >> 6,  d_ = n & 63;
        C[((size_t)((b_ * 8 + h_) * 64 + d_)) * 1024 + s_] = (__bf16)acc[j][r];
      }
  }
}

// ---------------------------------------------------------------------------
// 3) Attention: one block (4 waves) per (b, h, 16-row q block).
//    LDS: scores strip [16][SROW] f32 + 16 inv-rowsum floats (dynamic smem).
// ---------------------------------------------------------------------------
__global__ __launch_bounds__(128) void attn_kernel(
    const __bf16* __restrict__ Qb, const __bf16* __restrict__ Kb,
    const __bf16* __restrict__ Vt, __bf16* __restrict__ O) {
  extern __shared__ float smem[];
  float* sc   = smem;               // 16 * SROW
  float* rinv = smem + 16 * SROW;   // 16

  int wv = threadIdx.x >> 5, ln = threadIdx.x & 31;
  int l15 = ln & 15, half = ln >> 4;

  int bh = blockIdx.x >> 6, qt = blockIdx.x & 63;
  int b = bh >> 3, h = bh & 7;
  int q0 = qt * 16;

  // ---- Phase 1: scores = Q K^T (16 rows x 1024 keys) into LDS ----
  const __bf16* Qrow = Qb + (size_t)(b * 1024 + q0 + l15) * 512 + h * 64;
  v16bf a0 = load_a_bf16(Qrow,      half);   // k = 0..31 of dh
  v16bf a1 = load_a_bf16(Qrow + 32, half);   // k = 32..63

  for (int nt = wv; nt < 64; nt += 4) {
    int n0 = nt * 16;
    const __bf16* Krow = Kb + (size_t)(b * 1024 + n0 + l15) * 512 + h * 64;
    v16bf b0 = load_b_bf16(Krow,      half);
    v16bf b1 = load_b_bf16(Krow + 32, half);
    v8f cc = {};
    cc = wmma_bf16(a0, b0, cc);
    cc = wmma_bf16(a1, b1, cc);
#pragma unroll
    for (int r = 0; r < 8; ++r)
      sc[(r + half * 8) * SROW + n0 + l15] = cc[r];
  }
  __syncthreads();

  // ---- Phase 2: softmax over each of the 16 rows (scale folded in) ----
  {
    int row = threadIdx.x >> 3, sub = threadIdx.x & 7;   // 8 lanes per row
    float* srow = sc + row * SROW + sub * 128;
    float mx = -3.0e38f;
    for (int i = 0; i < 128; ++i) mx = fmaxf(mx, srow[i]);
#pragma unroll
    for (int d = 1; d < 8; d <<= 1) mx = fmaxf(mx, __shfl_xor(mx, d, 32));
    const float scale = 0.04419417382415922f;            // 1/sqrt(512)
    float sum = 0.f;
    for (int i = 0; i < 128; ++i) {
      float e = __expf((srow[i] - mx) * scale);
      srow[i] = e;                                       // unnormalized
      sum += e;
    }
#pragma unroll
    for (int d = 1; d < 8; d <<= 1) sum += __shfl_xor(sum, d, 32);
    if (sub == 0) rinv[row] = 1.0f / sum;
  }
  __syncthreads();

  // ---- Phase 3: O = P @ V (each wave owns one 16-wide dh tile) ----
  int n0 = wv * 16;
  const __bf16* Vrow = Vt + (size_t)((b * 8 + h) * 64 + n0 + l15) * 1024;
  float inv = rinv[l15];   // A-fragment row = lane&15 for all lanes

  v8f acc = {};
  for (int k0 = 0; k0 < 1024; k0 += 32) {
    if (k0 + 32 < 1024) __builtin_prefetch(Vrow + k0 + 32, 0, 0);
    const float* prow = sc + l15 * SROW + k0 + half * 8;
    v16bf a;
#pragma unroll
    for (int i = 0; i < 8; ++i) a[i]     = (__bf16)(prow[i]      * inv);
#pragma unroll
    for (int i = 0; i < 8; ++i) a[8 + i] = (__bf16)(prow[16 + i] * inv);
    v16bf bb = load_b_bf16(Vrow + k0, half);
    acc = wmma_bf16(a, bb, acc);
  }
#pragma unroll
  for (int r = 0; r < 8; ++r) {
    int m = b * 1024 + q0 + r + half * 8;
    O[(size_t)m * 512 + h * 64 + n0 + l15] = (__bf16)acc[r];
  }
}

// ---------------------------------------------------------------------------
// 4) Output projection: out(f32) = O(bf16) @ Wo + bo, 16x64 slab per wave
// ---------------------------------------------------------------------------
__global__ __launch_bounds__(128) void outproj_kernel(
    const __bf16* __restrict__ A, const __bf16* __restrict__ Wt,
    const float* __restrict__ bias, float* __restrict__ C) {
  int wv = threadIdx.x >> 5, ln = threadIdx.x & 31;
  int l15 = ln & 15, half = ln >> 4;
  int slab = blockIdx.x * 4 + wv;
  int mt = slab >> 3, ng = slab & 7;
  int m0 = mt * 16, n0 = ng * 64;

  const __bf16* Arow = A + (size_t)(m0 + l15) * 512;
  const __bf16* Brow[4];
  v8f acc[4];
#pragma unroll
  for (int j = 0; j < 4; ++j) {
    Brow[j] = Wt + (size_t)(n0 + j * 16 + l15) * 512;
    float bval = bias[n0 + j * 16 + l15];
#pragma unroll
    for (int r = 0; r < 8; ++r) acc[j][r] = bval;
  }

  for (int k0 = 0; k0 < 512; k0 += 32) {
    if (k0 + 32 < 512) {
      __builtin_prefetch(Arow + k0 + 32, 0, 0);
#pragma unroll
      for (int j = 0; j < 4; ++j) __builtin_prefetch(Brow[j] + k0 + 32, 0, 0);
    }
    v16bf a = load_a_bf16(Arow + k0, half);
#pragma unroll
    for (int j = 0; j < 4; ++j) {
      v16bf b = load_b_bf16(Brow[j] + k0, half);
      acc[j] = wmma_bf16(a, b, acc[j]);
    }
  }
#pragma unroll
  for (int j = 0; j < 4; ++j)
#pragma unroll
    for (int r = 0; r < 8; ++r) {
      int m = m0 + r + half * 8;
      C[(size_t)m * 512 + n0 + j * 16 + l15] = acc[j][r];
    }
}

// ---------------------------------------------------------------------------
extern "C" void kernel_launch(void* const* d_in, const int* in_sizes, int n_in,
                              void* d_out, int out_size, void* d_ws, size_t ws_size,
                              hipStream_t stream) {
  const float* q  = (const float*)d_in[0];
  const float* k  = (const float*)d_in[1];
  const float* v  = (const float*)d_in[2];
  const float* Wq = (const float*)d_in[3];
  const float* bq = (const float*)d_in[4];
  const float* Wk = (const float*)d_in[5];
  const float* bk = (const float*)d_in[6];
  const float* Wv = (const float*)d_in[7];
  const float* bv = (const float*)d_in[8];
  const float* Wo = (const float*)d_in[9];
  const float* bo = (const float*)d_in[10];

  char* ws = (char*)d_ws;
  const size_t WSZ = (size_t)512 * 512 * sizeof(__bf16);   // 512 KB per weight
  const size_t ASZ = (size_t)8192 * 512 * sizeof(__bf16);  // 8 MB per activation
  __bf16* WtQ = (__bf16*)(ws + 0 * WSZ);
  __bf16* WtK = (__bf16*)(ws + 1 * WSZ);
  __bf16* WtV = (__bf16*)(ws + 2 * WSZ);
  __bf16* WtO = (__bf16*)(ws + 3 * WSZ);
  __bf16* Qb  = (__bf16*)(ws + 4 * WSZ);
  __bf16* Kb  = (__bf16*)(ws + 4 * WSZ + 1 * ASZ);
  __bf16* Vt  = (__bf16*)(ws + 4 * WSZ + 2 * ASZ);
  __bf16* Ob  = (__bf16*)(ws + 4 * WSZ + 3 * ASZ);

  // 1) weight convert+transpose (512*512 elems each, 256 thr/blk)
  wconv_kernel<<<1024, 256, 0, stream>>>(Wq, WtQ);
  wconv_kernel<<<1024, 256, 0, stream>>>(Wk, WtK);
  wconv_kernel<<<1024, 256, 0, stream>>>(Wv, WtV);
  wconv_kernel<<<1024, 256, 0, stream>>>(Wo, WtO);

  // 2) projections: 512 row-tiles x 8 col-groups = 4096 slabs, 4/block
  proj_kernel<<<1024, 128, 0, stream>>>(q, WtQ, bq, Qb, 0);
  proj_kernel<<<1024, 128, 0, stream>>>(k, WtK, bk, Kb, 0);
  proj_kernel<<<1024, 128, 0, stream>>>(v, WtV, bv, Vt, 1);

  // 3) attention: B*H*(S/16) = 4096 blocks, dynamic LDS = scores + rowsums
  size_t shmem = (size_t)(16 * SROW + 16) * sizeof(float);  // ~65.9 KB
  attn_kernel<<<4096, 128, shmem, stream>>>(Qb, Kb, Vt, Ob);

  // 4) output projection -> d_out (f32)
  outproj_kernel<<<1024, 128, 0, stream>>>(Ob, WtO, bo, (float*)d_out);
}
